// PFMLPScaledXY_59176059404601
// MI455X (gfx1250) — matrix operand order repaired and checked by reading, not compiled, not run on hardware
//
#include <hip/hip_runtime.h>
#include <stdint.h>

typedef __attribute__((ext_vector_type(16))) __bf16 v16bf;
typedef __attribute__((ext_vector_type(8)))  float  v8f;

union Frag { uint4 q[2]; v16bf v; };

// ---------------- bf16 helpers (round-to-nearest-even) ----------------
__device__ __forceinline__ uint32_t f2bf_pk(float a, float b) {
  uint32_t ua = __builtin_bit_cast(uint32_t, a);
  uint32_t ub = __builtin_bit_cast(uint32_t, b);
  ua = ua + 0x7FFFu + ((ua >> 16) & 1u);
  ub = ub + 0x7FFFu + ((ub >> 16) & 1u);
  return (ua >> 16) | (ub & 0xFFFF0000u);
}
__device__ __forceinline__ uint16_t f2bf(float a) {
  uint32_t ua = __builtin_bit_cast(uint32_t, a);
  ua = ua + 0x7FFFu + ((ua >> 16) & 1u);
  return (uint16_t)(ua >> 16);
}

// ------ weight pre-pack: f32 [2][K][N] -> bf16 WMMA-B tiles -----------------
// Tile = 32(K) x 16(N) bf16 = 256 dwords. n-tiles grouped by NTB so all NTB*2
// tiles consumed per K-step form one contiguous (NTB*2) KB block:
//   tile linear index T = ((ng*KT + kt)*NTB + j)*2 + mode,  nt = ng*NTB + j
// Within tile, dword (l*8 + v) = pair (W[k], W[k+1]) with
//   k = kt*32 + (l & 16) + 2v,  n = nt*16 + (l & 15)
// so lane l reads its 8 B-fragment dwords as two contiguous b128 loads.
__global__ void pack_weights(const float* __restrict__ src,
                             uint32_t* __restrict__ dst, int K, int N, int NTB) {
  int t = blockIdx.x * blockDim.x + threadIdx.x;
  if (t >= K * N) return;                 // K*N dwords total (2 modes, bf16)
  int T = t >> 8;                         // tile index
  int w = t & 255;
  int l = w >> 3;                         // lane 0..31
  int v = w & 7;                          // B-fragment VGPR 0..7
  int mode = T & 1;
  int q    = T >> 1;
  int KT   = K >> 5;
  int j    = q % NTB;
  int q2   = q / NTB;
  int kt   = q2 % KT;
  int ng   = q2 / KT;
  int k = kt * 32 + (l & 16) + 2 * v;
  int n = (ng * NTB + j) * 16 + (l & 15);
  const float* s = src + (long)(mode * K + k) * N + n;
  dst[t] = f2bf_pk(s[0], s[N]);           // (k, n) and (k+1, n)
}

// ---------------- fragment loaders ------------------------------------------
__device__ __forceinline__ void load_a(Frag& f, const uint16_t* arow,
                                       int kt, int aoff) {
  f.q[0] = *(const uint4*)(arow + kt * 32 + aoff);
  f.q[1] = *(const uint4*)(arow + kt * 32 + aoff + 8);
}
template<int NTB>
__device__ __forceinline__ void load_b(Frag* fb, const uint32_t* wt, int kt) {
  const uint32_t* wk = wt + (size_t)kt * (NTB * 512);
#pragma unroll
  for (int j = 0; j < NTB; ++j) {
    fb[2 * j + 0].q[0] = *(const uint4*)(wk + j * 512 + 0);
    fb[2 * j + 0].q[1] = *(const uint4*)(wk + j * 512 + 4);
    fb[2 * j + 1].q[0] = *(const uint4*)(wk + j * 512 + 256);
    fb[2 * j + 1].q[1] = *(const uint4*)(wk + j * 512 + 260);
  }
}

// ---------------- fused phase-gated MLP layer -------------------------------
// A in LDS, row-major 64 x K bf16. Wave owns 16 rows (mb) and every other
// NTB-wide n-group (nsplit). One A fragment feeds NTB*2 WMMAs; fragments for
// K-step kt+1 are prefetched (double-buffered) while kt's 8 WMMAs execute,
// giving a full K-step of load-latency tolerance.
template<int K, int N, int NTB, bool ACT, bool TOG>
__device__ __forceinline__ void layer_run(
    const uint16_t* __restrict__ Ain, uint16_t* __restrict__ Aout,
    float* __restrict__ gout, const uint32_t* __restrict__ wpk,
    const float* __restrict__ bias, int mb, int lane, int nsplit, int rbaseC,
    const float* p0r, const float* p1r, long R) {
  constexpr int KT = K / 32;
  constexpr int NG = N / 16 / NTB;        // n-groups
  const uint16_t* arow = Ain + (mb + (lane & 15)) * K;
  const int aoff = (lane >> 4) << 4;      // halves: lanes>=16 start at K=16

  for (int g = nsplit; g < NG; g += 2) {
    v8f acc0[NTB];
    v8f acc1[NTB];
#pragma unroll
    for (int j = 0; j < NTB; ++j) { acc0[j] = {}; acc1[j] = {}; }

    const uint32_t* wt = wpk + (size_t)g * KT * NTB * 512 + lane * 8;

    Frag fa[2];
    Frag fb[2][NTB * 2];
    load_a(fa[0], arow, 0, aoff);
    load_b<NTB>(fb[0], wt, 0);

#pragma unroll 2
    for (int kt = 0; kt < KT - 1; ++kt) {
      const int cur = kt & 1, nxt = cur ^ 1;
      load_a(fa[nxt], arow, kt + 1, aoff);      // prefetch next K-step
      load_b<NTB>(fb[nxt], wt, kt + 1);
#pragma unroll
      for (int j = 0; j < NTB; ++j) {
        acc0[j] = __builtin_amdgcn_wmma_f32_16x16x32_bf16(
            false, fa[cur].v, false, fb[cur][2 * j + 0].v, (short)0, acc0[j],
            false, false);
        acc1[j] = __builtin_amdgcn_wmma_f32_16x16x32_bf16(
            false, fa[cur].v, false, fb[cur][2 * j + 1].v, (short)0, acc1[j],
            false, false);
      }
    }
    {                                           // peeled last K-step
      constexpr int cur = (KT - 1) & 1;
#pragma unroll
      for (int j = 0; j < NTB; ++j) {
        acc0[j] = __builtin_amdgcn_wmma_f32_16x16x32_bf16(
            false, fa[cur].v, false, fb[cur][2 * j + 0].v, (short)0, acc0[j],
            false, false);
        acc1[j] = __builtin_amdgcn_wmma_f32_16x16x32_bf16(
            false, fa[cur].v, false, fb[cur][2 * j + 1].v, (short)0, acc1[j],
            false, false);
      }
    }

#pragma unroll
    for (int j = 0; j < NTB; ++j) {
      const int n    = (g * NTB + j) * 16 + (lane & 15);
      const float b0 = bias[n];
      const float b1 = bias[N + n];
#pragma unroll
      for (int vg = 0; vg < 8; ++vg) {    // C/D layout: row = rbaseC + vg
        float d = p0r[vg] * (acc0[j][vg] + b0) + p1r[vg] * (acc1[j][vg] + b1);
        if (ACT) d = d > 0.0f ? d : (__expf(d) - 1.0f);  // ELU(alpha=1)
        if (TOG) gout[(R + rbaseC + vg) * (long)N + n] = d;
        else     Aout[(rbaseC + vg) * N + n] = f2bf(d);
      }
    }
  }
}

#define MT 64
__global__ __launch_bounds__(256, 1)
void pf_mlp(const float* __restrict__ x,  const float* __restrict__ b1,
            const float* __restrict__ b2, const float* __restrict__ b3,
            const uint32_t* __restrict__ pw1, const uint32_t* __restrict__ pw2,
            const uint32_t* __restrict__ pw3, float* __restrict__ out) {
  __shared__ uint16_t bufA[MT * 1024];    // 128 KB ping
  __shared__ uint16_t bufB[MT * 1024];    // 128 KB pong (CDNA5: 320KB/WGP)
  __shared__ float p0s[MT], p1s[MT];

  const int  tid = threadIdx.x;
  const long R   = (long)blockIdx.x * MT;

  // Stage phases + x[:, :256] -> bf16 in bufB (row stride 256)
  if (tid < MT) {
    p0s[tid] = x[(R + tid) * 258 + 256];
    p1s[tid] = x[(R + tid) * 258 + 257];
  }
  for (int i = tid; i < MT * 128; i += 256) {
    int r = i >> 7, c2 = i & 127;
    const float* s = x + (R + r) * 258 + c2 * 2;
    ((uint32_t*)bufB)[r * 128 + c2] = f2bf_pk(s[0], s[1]);
  }
  __syncthreads();

  const int lane   = tid & 31;
  const int wave   = tid >> 5;
  const int mb     = (wave & 3) * 16;     // 4 M-waves x 2 N-split waves
  const int nsplit = wave >> 2;
  const int rbaseC = mb + ((lane >> 4) << 3);

  float p0r[8], p1r[8];                   // per-lane phases for C/D rows
#pragma unroll
  for (int vg = 0; vg < 8; ++vg) {
    p0r[vg] = p0s[rbaseC + vg];
    p1r[vg] = p1s[rbaseC + vg];
  }

  layer_run<256, 1024, 4, true, false>(bufB, bufA, nullptr, pw1, b1,
                                       mb, lane, nsplit, rbaseC, p0r, p1r, R);
  __syncthreads();
  layer_run<1024, 1024, 4, true, false>(bufA, bufB, nullptr, pw2, b2,
                                        mb, lane, nsplit, rbaseC, p0r, p1r, R);
  __syncthreads();
  layer_run<1024, 64, 2, false, true>(bufB, nullptr, out, pw3, b3,
                                      mb, lane, nsplit, rbaseC, p0r, p1r, R);
}

extern "C" void kernel_launch(void* const* d_in, const int* in_sizes, int n_in,
                              void* d_out, int out_size, void* d_ws, size_t ws_size,
                              hipStream_t stream) {
  const float* x  = (const float*)d_in[0];
  const float* w1 = (const float*)d_in[1];
  const float* b1 = (const float*)d_in[2];
  const float* w2 = (const float*)d_in[3];
  const float* b2 = (const float*)d_in[4];
  const float* w3 = (const float*)d_in[5];
  const float* b3 = (const float*)d_in[6];
  float* out = (float*)d_out;

  // Workspace: packed bf16 weights, 5.25 MB total
  uint32_t* pw1 = (uint32_t*)d_ws;        //  512x1024 bf16 -> 262144 dwords
  uint32_t* pw2 = pw1 + 262144;           // 2048x1024 bf16 -> 1048576 dwords
  uint32_t* pw3 = pw2 + 1048576;          // 2048x  64 bf16 ->   65536 dwords

  pack_weights<<<262144 / 256, 256, 0, stream>>>(w1, pw1, 256, 1024, 4);
  pack_weights<<<1048576 / 256, 256, 0, stream>>>(w2, pw2, 1024, 1024, 4);
  pack_weights<<<65536 / 256, 256, 0, stream>>>(w3, pw3, 1024, 64, 2);

  pf_mlp<<<32768 / MT, 256, 0, stream>>>(x, b1, b2, b3, pw1, pw2, pw3, out);
}
